// ZImageTransformer2DModel_6588479832565
// MI455X (gfx1250) — matrix operand order repaired and checked by reading, not compiled, not run on hardware
//
#include <hip/hip_runtime.h>
#include <hip/hip_bf16.h>

// Performer / FAVOR+ linear attention for MI455X (gfx1250, wave32, WMMA + TDM).
// B=4, S=4096, H=16, D=64, M=128. Matmuls via v_wmma_f32_16x16x32_f16;
// bulk k/v tile traffic via tensor_load_to_lds (Tensor Data Mover).

#define BB   4
#define SS   4096
#define HH   16
#define DD   64
#define MM   128
#define EPSF 1e-6f
#define SCALE_X 0.35355339059327373f   // 64^-0.25
#define MINV    0.08838834764831845f   // 128^-0.5

#define SPLIT1 16   // colmax pass: s-split per (b,h)
#define SPLIT2 8    // kv pass:     s-split per (b,h)
#define NTHREADS 256

typedef __attribute__((ext_vector_type(16))) _Float16 v16h;
typedef __attribute__((ext_vector_type(8)))  float    v8f;

// TDM descriptor vector types (amdgpu-toolchain 6-arg builtin form)
typedef __attribute__((ext_vector_type(4))) unsigned int tdm_v4u;
typedef __attribute__((ext_vector_type(8))) int          tdm_v8i;
typedef __attribute__((ext_vector_type(4))) int          tdm_v4i;

// ---- monotone float<->uint encoding for atomic max on floats ----
__device__ __forceinline__ unsigned fkey(float f) {
    unsigned u = __float_as_uint(f);
    return (u & 0x80000000u) ? ~u : (u | 0x80000000u);
}
__device__ __forceinline__ float fdecode(unsigned k) {
    unsigned u = (k & 0x80000000u) ? (k ^ 0x80000000u) : ~k;
    return __uint_as_float(u);
}

// ---- WMMA f16 16x16x32 fragment index helpers (per ISA 7.12.2) ----
__device__ __forceinline__ int akidx(int lane, int i) {
    return ((i >> 3) << 4) + ((lane >> 4) << 3) + (i & 7);
}
__device__ __forceinline__ int bkidx(int lane, int i) {
    return ((i >> 3) << 4) + ((lane >> 4) << 3) + (i & 7);
}

// ---- TDM: DMA a rows x cols f32 tile (row stride in elements) into LDS ----
// D# per ISA 08_async_tensor §8: group0 = count/lds_addr/global_addr/type,
// group1 = data_size + tensor/tile dims + dim0 stride. 2D tensor -> higher groups zero.
__device__ __forceinline__ void tdm_load_tile_f32(const float* gsrc, unsigned lds_byte_off,
                                                  int rows, int cols, int row_stride_elems) {
    unsigned long long ga = (unsigned long long)(const void*)gsrc;
    tdm_v4u g0;
    g0[0] = 1u;                                   // count=1 valid descriptor
    g0[1] = lds_byte_off;                         // lds_addr
    g0[2] = (unsigned)ga;                         // global_addr[31:0]
    g0[3] = (unsigned)((ga >> 32) & 0x01FFFFFFu)  // global_addr[56:32]
            | 0x80000000u;                        // type=2 ("image") in bits 127:126
    tdm_v8i g1;
    g1[0] = 0x20000;                              // wgmask=0, data_size=2 (4 bytes)
    g1[1] = (int)((unsigned)cols << 16);          // tensor_dim0[15:0] in bits 63:48
    g1[2] = (int)((unsigned)rows << 16);          // tensor_dim1[15:0] in bits 95:80
    g1[3] = (int)((unsigned)cols << 16);          // tile_dim0 in bits 127:112
    g1[4] = rows;                                 // tile_dim1; tile_dim2=0 (unused)
    g1[5] = row_stride_elems;                     // tensor_dim0_stride[31:0]
    g1[6] = 0;
    g1[7] = 0;
    tdm_v4i gz4 = {0, 0, 0, 0};
    tdm_v8i gz8 = {0, 0, 0, 0, 0, 0, 0, 0};
    __builtin_amdgcn_tensor_load_to_lds(g0, g1, gz4, gz4, gz8, 0);
}

// =====================================================================
// Kernel 0: zero the workspace (colmax keys / ksum / kv all zero-init).
// =====================================================================
__global__ void favor_init_ws(unsigned* __restrict__ ws, int n) {
    int i = blockIdx.x * blockDim.x + threadIdx.x;
    if (i < n) ws[i] = 0u;
}

// =====================================================================
// Kernel 1: per-(b,h,m) max over s of (x_k . proj^T - 0.5||x_k||^2)
// =====================================================================
__global__ void favor_colmax(const float* __restrict__ kin,
                             const float* __restrict__ proj,
                             unsigned* __restrict__ colmax) {
    __shared__ _Float16 projl[MM * DD];   // 16 KB
    __shared__ _Float16 xl[32 * DD];      // 4 KB
    __shared__ float    rowsq[32];
    __shared__ unsigned cmax[MM];

    const int tid  = threadIdx.x;
    const int lane = tid & 31;
    const int wave = tid >> 5;
    const int bh   = blockIdx.x / SPLIT1;
    const int sb   = blockIdx.x % SPLIT1;
    const int b    = bh / HH, h = bh % HH;
    const int sbase = sb * (SS / SPLIT1);
    const int NCH  = (SS / SPLIT1) / 32;

    for (int i = tid; i < MM * DD; i += NTHREADS) projl[i] = (_Float16)proj[i];
    if (tid < MM) cmax[tid] = 0u;
    __syncthreads();

    for (int c = 0; c < NCH; ++c) {
        const int s0 = sbase + c * 32;
        for (int i = tid; i < 32 * DD; i += NTHREADS) {
            int r = i >> 6, dd = i & 63;
            size_t gi = ((size_t)(b * SS + s0 + r) * HH + h) * DD + dd;
            xl[i] = (_Float16)(kin[gi] * SCALE_X);
            if (c + 1 < NCH)
                __builtin_prefetch(&kin[gi + (size_t)32 * HH * DD], 0, 3);  // next chunk
        }
        __syncthreads();
        if (tid < 32) {
            float s = 0.f;
            for (int dd = 0; dd < DD; ++dd) {
                float xv = (float)xl[tid * DD + dd];
                s += xv * xv;
            }
            rowsq[tid] = 0.5f * s;
        }
        __syncthreads();

        // 32x128 projection: wave -> column tile, 2 row tiles.
        for (int rt = 0; rt < 2; ++rt) {
            const int row0 = rt * 16, col0 = wave * 16;
            v8f acc = {};
#pragma unroll
            for (int kb = 0; kb < DD; kb += 32) {
                v16h a, bf;
#pragma unroll
                for (int i = 0; i < 16; ++i)
                    a[i] = xl[(row0 + (lane & 15)) * DD + kb + akidx(lane, i)];
#pragma unroll
                for (int i = 0; i < 16; ++i)
                    bf[i] = projl[(col0 + (lane & 15)) * DD + kb + bkidx(lane, i)];
                acc = __builtin_amdgcn_wmma_f32_16x16x32_f16(
                    false, a, false, bf, (short)0, acc, false, false);
            }
#pragma unroll
            for (int j = 0; j < 8; ++j) {
                int r   = row0 + j + ((lane >> 4) << 3);
                int col = col0 + (lane & 15);
                float val = acc[j] - rowsq[r];
                atomicMax(&cmax[col], fkey(val));
            }
        }
        __syncthreads();
    }
    if (tid < MM) atomicMax(&colmax[bh * MM + tid], cmax[tid]);
}

// =====================================================================
// Kernel 2: k_phi = exp(proj - rowsq - colmax)*MINV + EPS (masked);
//           kv += k_phi^T v ; ksum += sum_s k_phi
//           k/v tiles staged via TDM (tensor_load_to_lds).
// =====================================================================
__global__ void favor_kv(const float* __restrict__ kin,
                         const float* __restrict__ vin,
                         const float* __restrict__ proj,
                         const unsigned char* __restrict__ mask,
                         const unsigned* __restrict__ colmax,
                         float* __restrict__ kv,
                         float* __restrict__ ksum) {
    __shared__ _Float16 projl[MM * DD];  // 16 KB
    __shared__ float    xstage[32 * DD]; // 8 KB  (TDM destination, raw k)
    __shared__ float    vstage[32 * DD]; // 8 KB  (TDM destination, raw v)
    __shared__ _Float16 xl[32 * DD];     // 4 KB
    __shared__ _Float16 vl[32 * DD];     // 4 KB
    __shared__ _Float16 kphi[32 * MM];   // 8 KB
    __shared__ float    rowsq[32];
    __shared__ float    maskrow[32];
    __shared__ float    cml[MM];

    const int tid  = threadIdx.x;
    const int lane = tid & 31;
    const int wave = tid >> 5;
    const int bh   = blockIdx.x / SPLIT2;
    const int sb   = blockIdx.x % SPLIT2;
    const int b    = bh / HH, h = bh % HH;
    const int sbase = sb * (SS / SPLIT2);

    // LDS byte offsets of the staging buffers (aperture keeps offset in addr[31:0])
    const unsigned xoff = (unsigned)(unsigned long long)(const void*)&xstage[0];
    const unsigned voff = (unsigned)(unsigned long long)(const void*)&vstage[0];

    for (int i = tid; i < MM * DD; i += NTHREADS) projl[i] = (_Float16)proj[i];
    if (tid < MM) cml[tid] = fdecode(colmax[bh * MM + tid]);
    __syncthreads();

    float ksum_acc = 0.f;         // valid for tid < MM
    v8f kvacc[4];                 // wave w accumulates kv rows m in [16w,16w+16), all d
#pragma unroll
    for (int t = 0; t < 4; ++t) kvacc[t] = (v8f){};

    for (int c = 0; c < (SS / SPLIT2) / 32; ++c) {
        const int s0 = sbase + c * 32;

        // TDM: DMA the 32x64 strided k and v tiles into LDS (wave 0 issues; EXEC ignored).
        if (wave == 0) {
            const float* kb0 = kin + ((size_t)(b * SS + s0) * HH + h) * DD;
            const float* vb0 = vin + ((size_t)(b * SS + s0) * HH + h) * DD;
            tdm_load_tile_f32(kb0, xoff, 32, DD, HH * DD);
            tdm_load_tile_f32(vb0, voff, 32, DD, HH * DD);
            __builtin_amdgcn_s_wait_tensorcnt(0);
        }
        if (tid < 32) maskrow[tid] = mask[b * SS + s0 + tid] ? 1.f : 0.f;
        __syncthreads();

        // f32 staging -> f16 operand tiles; rowsq from f32
        for (int i = tid; i < 32 * DD; i += NTHREADS) {
            xl[i] = (_Float16)(xstage[i] * SCALE_X);
            vl[i] = (_Float16)vstage[i];
        }
        if (tid < 32) {
            float s = 0.f;
            for (int dd = 0; dd < DD; ++dd) {
                float xv = xstage[tid * DD + dd] * SCALE_X;
                s += xv * xv;
            }
            rowsq[tid] = 0.5f * s;
        }
        __syncthreads();

        // projection + feature map -> kphi (LDS, f16)
        for (int rt = 0; rt < 2; ++rt) {
            const int row0 = rt * 16, col0 = wave * 16;
            v8f acc = {};
#pragma unroll
            for (int kb = 0; kb < DD; kb += 32) {
                v16h a, bf;
#pragma unroll
                for (int i = 0; i < 16; ++i)
                    a[i] = xl[(row0 + (lane & 15)) * DD + kb + akidx(lane, i)];
#pragma unroll
                for (int i = 0; i < 16; ++i)
                    bf[i] = projl[(col0 + (lane & 15)) * DD + kb + bkidx(lane, i)];
                acc = __builtin_amdgcn_wmma_f32_16x16x32_f16(
                    false, a, false, bf, (short)0, acc, false, false);
            }
#pragma unroll
            for (int j = 0; j < 8; ++j) {
                int r   = row0 + j + ((lane >> 4) << 3);
                int col = col0 + (lane & 15);
                float e = __expf(acc[j] - rowsq[r] - cml[col]) * MINV + EPSF;
                kphi[r * MM + col] = (_Float16)(e * maskrow[r]);
            }
        }
        __syncthreads();

        // ksum partial
        if (tid < MM) {
            float s = 0.f;
            for (int r = 0; r < 32; ++r) s += (float)kphi[r * MM + tid];
            ksum_acc += s;
        }
        // kv accumulate: A = kphi^T (16m x 32s), B = v (32s x 16d)
        {
            const int row0m = wave * 16;
#pragma unroll
            for (int dt = 0; dt < 4; ++dt) {
                v16h a, bf;
#pragma unroll
                for (int i = 0; i < 16; ++i) {
                    int ks = akidx(lane, i);
                    a[i] = kphi[ks * MM + row0m + (lane & 15)];
                }
#pragma unroll
                for (int i = 0; i < 16; ++i) {
                    int ks = bkidx(lane, i);
                    bf[i] = vl[ks * DD + dt * 16 + (lane & 15)];
                }
                kvacc[dt] = __builtin_amdgcn_wmma_f32_16x16x32_f16(
                    false, a, false, bf, (short)0, kvacc[dt], false, false);
            }
        }
        __syncthreads();
    }

    // commit
#pragma unroll
    for (int dt = 0; dt < 4; ++dt) {
#pragma unroll
        for (int j = 0; j < 8; ++j) {
            int m  = wave * 16 + j + ((lane >> 4) << 3);
            int dd = dt * 16 + (lane & 15);
            atomicAdd(&kv[((size_t)bh * MM + m) * DD + dd], kvacc[dt][j]);
        }
    }
    if (tid < MM) atomicAdd(&ksum[bh * MM + tid], ksum_acc);
}

// =====================================================================
// Kernel 3: q_phi (row-max stabilized), out = q_phi kv / (q_phi ksum + eps)
// =====================================================================
__global__ void favor_out(const float* __restrict__ qin,
                          const float* __restrict__ proj,
                          const unsigned char* __restrict__ mask,
                          const float* __restrict__ kv,
                          const float* __restrict__ ksum,
                          float* __restrict__ out) {
    __shared__ _Float16 projl[MM * DD];  // 16 KB
    __shared__ _Float16 kvl[MM * DD];    // 16 KB
    __shared__ _Float16 xl[32 * DD];     // 4 KB
    __shared__ _Float16 qphi[32 * MM];   // 8 KB
    __shared__ float    ksl[MM];
    __shared__ float    rowsq[32];
    __shared__ unsigned rowmaxk[32];
    __shared__ float    denom[32];
    __shared__ float    maskrow[32];

    const int tid  = threadIdx.x;
    const int lane = tid & 31;
    const int wave = tid >> 5;
    const int bh   = blockIdx.x / (SS / 32);
    const int sblk = blockIdx.x % (SS / 32);
    const int b    = bh / HH, h = bh % HH;
    const int s0   = sblk * 32;

    for (int i = tid; i < MM * DD; i += NTHREADS) {
        projl[i] = (_Float16)proj[i];
        kvl[i]   = (_Float16)kv[(size_t)bh * MM * DD + i];
    }
    if (tid < MM) ksl[tid] = ksum[bh * MM + tid];
    for (int i = tid; i < 32 * DD; i += NTHREADS) {
        int r = i >> 6, dd = i & 63;
        xl[i] = (_Float16)(qin[((size_t)(b * SS + s0 + r) * HH + h) * DD + dd] * SCALE_X);
    }
    if (tid < 32) {
        rowmaxk[tid] = 0u;
        maskrow[tid] = mask[b * SS + s0 + tid] ? 1.f : 0.f;
    }
    __syncthreads();
    if (tid < 32) {
        float s = 0.f;
        for (int dd = 0; dd < DD; ++dd) {
            float xv = (float)xl[tid * DD + dd];
            s += xv * xv;
        }
        rowsq[tid] = 0.5f * s;
    }
    __syncthreads();

    // q projection; keep raw values in registers, publish per-row max via LDS atomics
    v8f pacc[2];
    for (int rt = 0; rt < 2; ++rt) {
        const int row0 = rt * 16, col0 = wave * 16;
        v8f acc = {};
#pragma unroll
        for (int kb = 0; kb < DD; kb += 32) {
            v16h a, bf;
#pragma unroll
            for (int i = 0; i < 16; ++i)
                a[i] = xl[(row0 + (lane & 15)) * DD + kb + akidx(lane, i)];
#pragma unroll
            for (int i = 0; i < 16; ++i)
                bf[i] = projl[(col0 + (lane & 15)) * DD + kb + bkidx(lane, i)];
            acc = __builtin_amdgcn_wmma_f32_16x16x32_f16(
                false, a, false, bf, (short)0, acc, false, false);
        }
#pragma unroll
        for (int j = 0; j < 8; ++j) {
            int r = row0 + j + ((lane >> 4) << 3);
            float val = acc[j] - rowsq[r];
            acc[j] = val;
            atomicMax(&rowmaxk[r], fkey(val));
        }
        pacc[rt] = acc;
    }
    __syncthreads();

    // q_phi -> LDS f16
    for (int rt = 0; rt < 2; ++rt) {
        const int row0 = rt * 16, col0 = wave * 16;
#pragma unroll
        for (int j = 0; j < 8; ++j) {
            int r   = row0 + j + ((lane >> 4) << 3);
            int col = col0 + (lane & 15);
            float e = __expf(pacc[rt][j] - fdecode(rowmaxk[r])) * MINV + EPSF;
            qphi[r * MM + col] = (_Float16)e;
        }
    }
    __syncthreads();

    if (tid < 32) {
        float s = 0.f;
        for (int m = 0; m < MM; ++m) s += (float)qphi[tid * MM + m] * ksl[m];
        denom[tid] = s;
    }
    __syncthreads();

    // out = q_phi (32x128) @ kv (128x64): 8 tiles, one per wave; K=128 -> 4 WMMAs
    {
        const int rt = wave >> 2, dt = wave & 3;
        const int row0 = rt * 16, d0 = dt * 16;
        v8f acc = {};
#pragma unroll
        for (int kb = 0; kb < MM; kb += 32) {
            v16h a, bf;
#pragma unroll
            for (int i = 0; i < 16; ++i)
                a[i] = qphi[(row0 + (lane & 15)) * MM + kb + akidx(lane, i)];
#pragma unroll
            for (int i = 0; i < 16; ++i)
                bf[i] = kvl[(kb + bkidx(lane, i)) * DD + d0 + (lane & 15)];
            acc = __builtin_amdgcn_wmma_f32_16x16x32_f16(
                false, a, false, bf, (short)0, acc, false, false);
        }
#pragma unroll
        for (int j = 0; j < 8; ++j) {
            int r  = row0 + j + ((lane >> 4) << 3);
            int dd = d0 + (lane & 15);
            float o = acc[j] / (denom[r] + EPSF) * maskrow[r];
            out[((size_t)(b * SS + s0 + r) * HH + h) * DD + dd] = o;
        }
    }
}

// =====================================================================
extern "C" void kernel_launch(void* const* d_in, const int* in_sizes, int n_in,
                              void* d_out, int out_size, void* d_ws, size_t ws_size,
                              hipStream_t stream) {
    const float* q    = (const float*)d_in[0];
    const float* k    = (const float*)d_in[1];
    const float* v    = (const float*)d_in[2];
    const float* proj = (const float*)d_in[3];
    const unsigned char* mask = (const unsigned char*)d_in[4];
    float* out = (float*)d_out;

    // workspace layout (dwords): colmax keys [B*H*M] | ksum [B*H*M] | kv [B*H*M*D]
    const int n_cm   = BB * HH * MM;           // 8192
    const int n_ks   = BB * HH * MM;           // 8192
    const int n_kv   = BB * HH * MM * DD;      // 524288
    const int n_ws   = n_cm + n_ks + n_kv;     // 540672 dwords (~2.1 MB)
    unsigned* colmax = (unsigned*)d_ws;
    float*    ksum   = (float*)d_ws + n_cm;
    float*    kvbuf  = (float*)d_ws + n_cm + n_ks;

    favor_init_ws<<<(n_ws + NTHREADS - 1) / NTHREADS, NTHREADS, 0, stream>>>((unsigned*)d_ws, n_ws);
    favor_colmax<<<BB * HH * SPLIT1, NTHREADS, 0, stream>>>(k, proj, colmax);
    favor_kv<<<BB * HH * SPLIT2, NTHREADS, 0, stream>>>(k, v, proj, mask, colmax, kvbuf, ksum);
    favor_out<<<BB * HH * (SS / 32), NTHREADS, 0, stream>>>(q, proj, mask, kvbuf, ksum, out);
}
